// SelfAttentionBlock_34866544509744
// MI455X (gfx1250) — compile-verified, matrix operand order checked by reading, not run on hardware
//
#include <hip/hip_runtime.h>
#include <hip/hip_bf16.h>

typedef __bf16 v16bf __attribute__((ext_vector_type(16)));
typedef __bf16 v8bf  __attribute__((ext_vector_type(8)));
typedef float  v8f   __attribute__((ext_vector_type(8)));

// ---------------------------------------------------------------------------
// WMMA fragment helpers (bf16 16x16x32, f32 accumulate).
// 16-bit A/B layout (CDNA5 ISA 7.12.2): lane l holds row (l&15); K components
// j<8 -> K = koff+j, j>=8 -> K = koff+8+j, koff = (l>=16)?8:0, i.e. two
// contiguous 8-element (16B) chunks at +0 and +16 within the 32-wide K tile.
// ---------------------------------------------------------------------------
__device__ __forceinline__ v16bf load_frag_rm(const __bf16* base, int stride,
                                              int r0, int k0) {
  int l = threadIdx.x & 31;
  const __bf16* p = base + (r0 + (l & 15)) * stride + k0 + ((l >> 4) << 3);
  v8bf lo = *(const v8bf*)p;
  v8bf hi = *(const v8bf*)(p + 16);
  v16bf r;
#pragma unroll
  for (int i = 0; i < 8; ++i) { r[i] = lo[i]; r[i + 8] = hi[i]; }
  return r;
}

__device__ __forceinline__ v16bf load_frag_f32(const float* base, int stride,
                                               int r0, int k0) {
  int l = threadIdx.x & 31;
  const float* p = base + (r0 + (l & 15)) * stride + k0 + ((l >> 4) << 3);
  v16bf r;
#pragma unroll
  for (int i = 0; i < 8; ++i) { r[i] = (__bf16)p[i]; r[i + 8] = (__bf16)p[i + 16]; }
  return r;
}

// Pre-swizzled weight fragments: frag (nt,kt) = 32 lanes x 16 bf16 contiguous.
__device__ __forceinline__ v16bf load_frag_w(const __bf16* w, int KT, int nt, int kt) {
  const __bf16* p = w + ((size_t)((nt * KT + kt) * 32 + (threadIdx.x & 31))) * 16;
  v8bf lo = *(const v8bf*)p;
  v8bf hi = *(const v8bf*)(p + 8);
  v16bf r;
#pragma unroll
  for (int i = 0; i < 8; ++i) { r[i] = lo[i]; r[i + 8] = hi[i]; }
  return r;
}

__device__ __forceinline__ v8f wmma_bf16(v16bf a, v16bf b, v8f c) {
  return __builtin_amdgcn_wmma_f32_16x16x32_bf16(false, a, false, b, (short)0, c,
                                                 false, false);
}

// Async HBM->LDS copy, 16 bytes per lane (GLOBAL_LOAD_ASYNC_TO_LDS_B128,
// GVS mode: LDS-addr VGPR, 32-bit byte-offset VGPR, 64-bit SGPR base).
__device__ __forceinline__ void async_load_b128(void* lds_ptr,
                                                unsigned goffset_bytes,
                                                const void* base) {
  unsigned lds_addr = (unsigned)(unsigned long long)lds_ptr;
  asm volatile("global_load_async_to_lds_b128 %0, %1, %2"
               :
               : "v"(lds_addr), "v"(goffset_bytes),
                 "s"((unsigned long long)base)
               : "memory");
}

__device__ __forceinline__ void wait_asynccnt0() {
  asm volatile("s_wait_asynccnt 0" ::: "memory");
}

// ---------------------------------------------------------------------------
// Kernel: swizzle fp32 weight [N][K] into bf16 B-fragment layout.
// ---------------------------------------------------------------------------
__global__ void prep_w_kernel(const float* __restrict__ src, __bf16* __restrict__ dst,
                              int N, int K) {
  int g = blockIdx.x * 256 + threadIdx.x;
  if (g >= N * K) return;
  int KT = K >> 5;
  int frag = g >> 9;          // 512 elements per fragment
  int l = (g >> 4) & 31;
  int j = g & 15;
  int nt = frag / KT, kt = frag % KT;
  int n = nt * 16 + (l & 15);
  int k = kt * 32 + ((j < 8) ? j : j + 8) + ((l >= 16) ? 8 : 0);
  dst[g] = (__bf16)src[n * K + k];
}

// ---------------------------------------------------------------------------
// LN1: x[4,65536,192] -> nx bf16 (also consumed by LCE branch). 1 wave / row.
// ---------------------------------------------------------------------------
__global__ __launch_bounds__(256) void ln1_kernel(const float* __restrict__ x,
                                                  const float* __restrict__ g,
                                                  const float* __restrict__ b,
                                                  __bf16* __restrict__ nx) {
  int wid = threadIdx.x >> 5, lane = threadIdx.x & 31;
  size_t row = (size_t)blockIdx.x * 8 + wid;
  const float* xr = x + row * 192;
  float s = 0.f;
  for (int c = lane; c < 192; c += 32) s += xr[c];
#pragma unroll
  for (int o = 16; o > 0; o >>= 1) s += __shfl_xor(s, o, 32);
  float mean = s * (1.f / 192.f);
  float v = 0.f;
  for (int c = lane; c < 192; c += 32) { float d = xr[c] - mean; v += d * d; }
#pragma unroll
  for (int o = 16; o > 0; o >>= 1) v += __shfl_xor(v, o, 32);
  float inv = rsqrtf(v * (1.f / 192.f) + 1e-5f);
  __bf16* nr = nx + row * 192;
  for (int c = lane; c < 192; c += 32)
    nr[c] = (__bf16)((xr[c] - mean) * inv * g[c] + b[c]);
}

// ---------------------------------------------------------------------------
// Shifted-window attention. One block (8 waves) per 8x8 window.
// LDS: win/out 24K | q 24K | k 24K | vT 24K | attn 96K = 192K dynamic.
// ---------------------------------------------------------------------------
__device__ __forceinline__ int band(int p) { return p < 248 ? 0 : (p < 252 ? 1 : 2); }

__global__ __launch_bounds__(256) void attn_kernel(
    const float* __restrict__ x, const __bf16* __restrict__ nx,
    const __bf16* __restrict__ wqkv, const float* __restrict__ b_qkv,
    const float* __restrict__ rel_table, const __bf16* __restrict__ wproj,
    const float* __restrict__ b_proj, float* __restrict__ x1) {
  extern __shared__ char smem[];
  __bf16* s_win = (__bf16*)smem;                 // 64x192, reused for attn out
  __bf16* s_q   = (__bf16*)(smem + 24576);       // 64x192 (scaled)
  __bf16* s_k   = (__bf16*)(smem + 49152);       // 64x192
  __bf16* s_vT  = (__bf16*)(smem + 73728);       // [192][64]: (h*32+d)*64+j
  float*  s_at  = (float*)(smem + 98304);        // [6][64][64]

  int wno = blockIdx.x;
  int bimg = wno >> 10;
  int wy = (wno >> 5) & 31, wx = wno & 31;
  int tid = threadIdx.x, wid = tid >> 5, lane = tid & 31;

  // gather shifted window (roll by -SHIFT => src = pos+4 mod 256) via async
  // HBM->LDS copies (no VGPR round-trip), 16B per lane per issue.
#pragma unroll
  for (int it = 0; it < 6; ++it) {
    int idx = it * 256 + tid;               // < 64*24 chunks of 8 bf16
    int t = idx / 24, ch = (idx % 24) * 8;
    int gy = (wy * 8 + (t >> 3) + 4) & 255;
    int gx = (wx * 8 + (t & 7) + 4) & 255;
    unsigned goff =
        (unsigned)((((unsigned)bimg * 65536u + gy * 256 + gx) * 192u + ch) * 2u);
    async_load_b128(s_win + t * 192 + ch, goff, nx);
  }
  wait_asynccnt0();
  __syncthreads();

  // qkv GEMM: 64x192 @ 192x576, MT=4 NT=36 KT=6
  const float scale = 0.1767766953f;
  for (int t = wid; t < 4 * 36; t += 8) {
    int mt = t / 36, nt = t % 36;
    v8f acc = {};
#pragma unroll
    for (int kt = 0; kt < 6; ++kt)
      acc = wmma_bf16(load_frag_rm(s_win, 192, mt * 16, kt * 32),
                      load_frag_w(wqkv, 6, nt, kt), acc);
    int n = nt * 16 + (lane & 15);
    int mb = mt * 16 + ((lane >> 4) << 3);
    float bias = b_qkv[n];
    int ntu = __builtin_amdgcn_readfirstlane(nt);  // wave-uniform tile id
    if (ntu < 12) {
#pragma unroll
      for (int r = 0; r < 8; ++r)
        s_q[(mb + r) * 192 + n] = (__bf16)((acc[r] + bias) * scale);
    } else if (ntu < 24) {
#pragma unroll
      for (int r = 0; r < 8; ++r)
        s_k[(mb + r) * 192 + (n - 192)] = (__bf16)(acc[r] + bias);
    } else {
#pragma unroll
      for (int r = 0; r < 8; ++r)
        s_vT[(n - 384) * 64 + mb + r] = (__bf16)(acc[r] + bias);
    }
  }
  __syncthreads();

  // QK^T + rel-pos bias + shift mask: per head MT=4 NT=4 (K=32)
  for (int t = wid; t < 6 * 16; t += 8) {
    int h = t >> 4, mt = (t >> 2) & 3, nt = t & 3;
    v8f acc = {};
    acc = wmma_bf16(load_frag_rm(s_q, 192, mt * 16, h * 32),
                    load_frag_rm(s_k, 192, nt * 16, h * 32), acc);
    int j = nt * 16 + (lane & 15);
    int mb = mt * 16 + ((lane >> 4) << 3);
    int jy = j >> 3, jx = j & 7;
    int lbj = band(wy * 8 + jy) * 3 + band(wx * 8 + jx);
#pragma unroll
    for (int r = 0; r < 8; ++r) {
      int i = mb + r;
      int iy = i >> 3, ix = i & 7;
      int rel = (iy - jy + 7) * 15 + (ix - jx + 7);
      int lbi = band(wy * 8 + iy) * 3 + band(wx * 8 + ix);
      float m = (lbi != lbj) ? -100.f : 0.f;
      s_at[(h * 64 + i) * 64 + j] = acc[r] + rel_table[rel * 6 + h] + m;
    }
  }
  __syncthreads();

  // softmax over 384 rows of 64
  for (int rr = tid; rr < 384; rr += 256) {
    float* row = s_at + rr * 64;
    float mx = row[0];
    for (int j = 1; j < 64; ++j) mx = fmaxf(mx, row[j]);
    float s = 0.f;
    for (int j = 0; j < 64; ++j) { float e = __expf(row[j] - mx); row[j] = e; s += e; }
    float inv = 1.f / s;
    for (int j = 0; j < 64; ++j) row[j] *= inv;
  }
  __syncthreads();

  // attn @ V: per head MT=4 NT=2 KT=2 -> out bf16 into s_win
  for (int t = wid; t < 48; t += 8) {
    int h = t >> 3, mt = (t >> 1) & 3, nt = t & 1;
    v8f acc = {};
#pragma unroll
    for (int kt = 0; kt < 2; ++kt)
      acc = wmma_bf16(load_frag_f32(s_at + h * 4096, 64, mt * 16, kt * 32),
                      load_frag_rm(s_vT, 64, h * 32 + nt * 16, kt * 32), acc);
    int n = h * 32 + nt * 16 + (lane & 15);
    int mb = mt * 16 + ((lane >> 4) << 3);
#pragma unroll
    for (int r = 0; r < 8; ++r) s_win[(mb + r) * 192 + n] = (__bf16)acc[r];
  }
  __syncthreads();

  // proj + reverse shift + residual -> x1
  for (int t = wid; t < 48; t += 8) {
    int mt = t / 12, nt = t % 12;
    v8f acc = {};
#pragma unroll
    for (int kt = 0; kt < 6; ++kt)
      acc = wmma_bf16(load_frag_rm(s_win, 192, mt * 16, kt * 32),
                      load_frag_w(wproj, 6, nt, kt), acc);
    int n = nt * 16 + (lane & 15);
    int mb = mt * 16 + ((lane >> 4) << 3);
    float bp = b_proj[n];
#pragma unroll
    for (int r = 0; r < 8; ++r) {
      int tok = mb + r;
      int dy = (wy * 8 + (tok >> 3) + 4) & 255;
      int dx = (wx * 8 + (tok & 7) + 4) & 255;
      size_t di = ((size_t)bimg * 65536 + dy * 256 + dx) * 192 + n;
      x1[di] = x[di] + acc[r] + bp;
    }
  }
}

// ---------------------------------------------------------------------------
// LN2 + fused MLP (fc1+GELU+fc2) per 64-row tile. LDS: n2 24K + hid 96K.
// ---------------------------------------------------------------------------
__global__ __launch_bounds__(256) void mlp_kernel(
    const float* __restrict__ x1, const float* __restrict__ g2,
    const float* __restrict__ b2, const __bf16* __restrict__ wfc1,
    const float* __restrict__ b_fc1, const __bf16* __restrict__ wfc2,
    const float* __restrict__ b_fc2, float* __restrict__ out) {
  extern __shared__ char smem[];
  __bf16* s_n2 = (__bf16*)smem;             // 64x192
  __bf16* s_h  = (__bf16*)(smem + 24576);   // 64x768
  size_t row0 = (size_t)blockIdx.x * 64;
  int tid = threadIdx.x, wid = tid >> 5, lane = tid & 31;

  for (int r = wid; r < 64; r += 8) {
    const float* xr = x1 + (row0 + r) * 192;
    float s = 0.f;
    for (int c = lane; c < 192; c += 32) s += xr[c];
#pragma unroll
    for (int o = 16; o > 0; o >>= 1) s += __shfl_xor(s, o, 32);
    float mean = s * (1.f / 192.f);
    float v = 0.f;
    for (int c = lane; c < 192; c += 32) { float d = xr[c] - mean; v += d * d; }
#pragma unroll
    for (int o = 16; o > 0; o >>= 1) v += __shfl_xor(v, o, 32);
    float inv = rsqrtf(v * (1.f / 192.f) + 1e-5f);
    for (int c = lane; c < 192; c += 32)
      s_n2[r * 192 + c] = (__bf16)((xr[c] - mean) * inv * g2[c] + b2[c]);
  }
  __syncthreads();

  // fc1 + GELU: MT=4 NT=48 KT=6
  for (int t = wid; t < 192; t += 8) {
    int mt = t / 48, nt = t % 48;
    v8f acc = {};
#pragma unroll
    for (int kt = 0; kt < 6; ++kt)
      acc = wmma_bf16(load_frag_rm(s_n2, 192, mt * 16, kt * 32),
                      load_frag_w(wfc1, 6, nt, kt), acc);
    int n = nt * 16 + (lane & 15);
    int mb = mt * 16 + ((lane >> 4) << 3);
    float bb = b_fc1[n];
#pragma unroll
    for (int r = 0; r < 8; ++r) {
      float u = acc[r] + bb;
      float gl = 0.5f * u * (1.f + erff(u * 0.70710678f));
      s_h[(mb + r) * 768 + n] = (__bf16)gl;
    }
  }
  __syncthreads();

  // fc2 + residual: MT=4 NT=12 KT=24
  for (int t = wid; t < 48; t += 8) {
    int mt = t / 12, nt = t % 12;
    v8f acc = {};
#pragma unroll
    for (int kt = 0; kt < 24; ++kt)
      acc = wmma_bf16(load_frag_rm(s_h, 768, mt * 16, kt * 32),
                      load_frag_w(wfc2, 24, nt, kt), acc);
    int n = nt * 16 + (lane & 15);
    int mb = mt * 16 + ((lane >> 4) << 3);
    float bb = b_fc2[n];
#pragma unroll
    for (int r = 0; r < 8; ++r) {
      size_t i = (row0 + mb + r) * 192 + n;
      out[i] = x1[i] + acc[r] + bb;
    }
  }
}

// ---------------------------------------------------------------------------
// LCE branch (tiny FLOPs, VALU kernels).
// ---------------------------------------------------------------------------
__global__ void zero_kernel(float* p, int n) {
  int g = blockIdx.x * 256 + threadIdx.x;
  if (g < n) p[g] = 0.f;
}

__global__ void conv1_kernel(const __bf16* __restrict__ nx, const float* __restrict__ w,
                             const float* __restrict__ b, float* __restrict__ t1) {
  size_t g = (size_t)blockIdx.x * 256 + threadIdx.x;  // < 262144*24
  int o = (int)(g % 24);
  size_t p = g / 24;
  const __bf16* src = nx + p * 192;
  const float* wr = w + o * 192;
  float s = b[o];
  for (int c = 0; c < 192; ++c) s += (float)src[c] * wr[c];
  t1[p * 24 + o] = s;
}

__global__ void conv2_kernel(const float* __restrict__ t1, const float* __restrict__ w,
                             const float* __restrict__ b, float* __restrict__ t2) {
  size_t g = (size_t)blockIdx.x * 256 + threadIdx.x;
  int o = (int)(g % 24);
  size_t p = g / 24;
  int bi = (int)(p >> 16), pix = (int)(p & 65535);
  int y = pix >> 8, xc = pix & 255;
  float s = b[o];
  for (int dy = -1; dy <= 1; ++dy) {
    int yy = y + dy;
    if ((unsigned)yy >= 256u) continue;
    for (int dx = -1; dx <= 1; ++dx) {
      int xx = xc + dx;
      if ((unsigned)xx >= 256u) continue;
      const float* src = t1 + ((size_t)(bi << 16) + yy * 256 + xx) * 24;
      const float* wr = w + o * 216 + (dy + 1) * 3 + (dx + 1);
      for (int i = 0; i < 24; ++i) s += src[i] * wr[i * 9];
    }
  }
  t2[p * 24 + o] = s;
}

// conv3 (1x1, 24->192) + LeakyReLU + global-mean partial sums (atomic).
__global__ __launch_bounds__(192) void conv3_pool_kernel(
    const float* __restrict__ t2, const float* __restrict__ w3,
    const float* __restrict__ b3, float* __restrict__ pooled) {
  __shared__ float tile[128 * 24];
  size_t p0 = (size_t)blockIdx.x * 128;
  int bi = (int)(p0 >> 16);
  for (int i = threadIdx.x; i < 128 * 24; i += 192) tile[i] = t2[p0 * 24 + i];
  __syncthreads();
  int c = threadIdx.x;
  const float* wr = w3 + c * 24;
  float bb = b3[c], acc = 0.f;
  for (int p = 0; p < 128; ++p) {
    float s = bb;
    for (int r = 0; r < 24; ++r) s += tile[p * 24 + r] * wr[r];
    acc += (s >= 0.f) ? s : 0.2f * s;
  }
  atomicAdd(&pooled[bi * 192 + c], acc * (1.f / 65536.f));
}

__global__ __launch_bounds__(192) void se_kernel(const float* __restrict__ pooled,
                                                 const float* __restrict__ w1,
                                                 const float* __restrict__ w2,
                                                 float* __restrict__ wgt) {
  __shared__ float hid[24];
  for (int b = 0; b < 4; ++b) {
    if (threadIdx.x < 24) {
      float s = 0.f;
      for (int c = 0; c < 192; ++c) s += pooled[b * 192 + c] * w1[threadIdx.x * 192 + c];
      hid[threadIdx.x] = fmaxf(s, 0.f);
    }
    __syncthreads();
    float s = 0.f;
    for (int r = 0; r < 24; ++r) s += hid[r] * w2[threadIdx.x * 24 + r];
    wgt[b * 192 + threadIdx.x] = 1.f / (1.f + __expf(-s));
    __syncthreads();
  }
}

__global__ __launch_bounds__(192) void lce_final_kernel(
    const float* __restrict__ t2, const float* __restrict__ w3,
    const float* __restrict__ b3, const float* __restrict__ wgt,
    float* __restrict__ out) {
  __shared__ float tile[128 * 24];
  size_t p0 = (size_t)blockIdx.x * 128;
  int bi = (int)(p0 >> 16);
  for (int i = threadIdx.x; i < 128 * 24; i += 192) tile[i] = t2[p0 * 24 + i];
  __syncthreads();
  int c = threadIdx.x;
  const float* wr = w3 + c * 24;
  float bb = b3[c], wg = wgt[bi * 192 + c];
  for (int p = 0; p < 128; ++p) {
    float s = bb;
    for (int r = 0; r < 24; ++r) s += tile[p * 24 + r] * wr[r];
    float y = (s >= 0.f) ? s : 0.2f * s;
    out[(p0 + p) * 192 + c] += y * wg;
  }
}

// ---------------------------------------------------------------------------
// Host launch
// ---------------------------------------------------------------------------
extern "C" void kernel_launch(void* const* d_in, const int* in_sizes, int n_in,
                              void* d_out, int out_size, void* d_ws, size_t ws_size,
                              hipStream_t stream) {
  const float* x         = (const float*)d_in[0];
  const float* ln1_g     = (const float*)d_in[1];
  const float* ln1_b     = (const float*)d_in[2];
  const float* w_qkv     = (const float*)d_in[3];
  const float* b_qkv     = (const float*)d_in[4];
  const float* rel_table = (const float*)d_in[5];
  const float* w_proj    = (const float*)d_in[6];
  const float* b_proj    = (const float*)d_in[7];
  const float* ln2_g     = (const float*)d_in[8];
  const float* ln2_b     = (const float*)d_in[9];
  const float* w_fc1     = (const float*)d_in[10];
  const float* b_fc1     = (const float*)d_in[11];
  const float* w_fc2     = (const float*)d_in[12];
  const float* b_fc2     = (const float*)d_in[13];
  const float* conv1_w   = (const float*)d_in[14];
  const float* conv1_b   = (const float*)d_in[15];
  const float* conv2_w   = (const float*)d_in[16];
  const float* conv2_b   = (const float*)d_in[17];
  const float* conv3_w   = (const float*)d_in[18];
  const float* conv3_b   = (const float*)d_in[19];
  const float* se_w1     = (const float*)d_in[20];
  const float* se_w2     = (const float*)d_in[21];

  char* ws = (char*)d_ws;
  __bf16* nx     = (__bf16*)(ws + 0);               // 100663296 B
  float*  x1     = (float*)(ws + 100663296);        // 201326592 B
  float*  t1     = (float*)(ws + 301989888);        // 25165824 B
  float*  t2     = (float*)(ws + 327155712);        // 25165824 B
  float*  pooled = (float*)(ws + 352321536);        // 3072 B
  float*  wgt    = (float*)(ws + 352324608);        // 3072 B
  __bf16* wqkv_f = (__bf16*)(ws + 352327680);       // 221184 B
  __bf16* wproj_f= (__bf16*)(ws + 352548864);       // 73728 B
  __bf16* wfc1_f = (__bf16*)(ws + 352622592);       // 294912 B
  __bf16* wfc2_f = (__bf16*)(ws + 352917504);       // 294912 B

  // weight swizzle to bf16 fragments
  prep_w_kernel<<<432, 256, 0, stream>>>(w_qkv, wqkv_f, 576, 192);
  prep_w_kernel<<<144, 256, 0, stream>>>(w_proj, wproj_f, 192, 192);
  prep_w_kernel<<<576, 256, 0, stream>>>(w_fc1, wfc1_f, 768, 192);
  prep_w_kernel<<<576, 256, 0, stream>>>(w_fc2, wfc2_f, 192, 768);

  zero_kernel<<<3, 256, 0, stream>>>(pooled, 768);

  // LN1 -> nx (bf16)
  ln1_kernel<<<32768, 256, 0, stream>>>(x, ln1_g, ln1_b, nx);

  // shifted-window attention -> x1
  attn_kernel<<<4096, 256, 196608, stream>>>(x, nx, wqkv_f, b_qkv, rel_table,
                                             wproj_f, b_proj, x1);

  // LN2 + MLP -> out (x2)
  mlp_kernel<<<4096, 256, 122880, stream>>>(x1, ln2_g, ln2_b, wfc1_f, b_fc1,
                                            wfc2_f, b_fc2, (float*)d_out);

  // LCE branch
  conv1_kernel<<<24576, 256, 0, stream>>>(nx, conv1_w, conv1_b, t1);
  conv2_kernel<<<24576, 256, 0, stream>>>(t1, conv2_w, conv2_b, t2);
  conv3_pool_kernel<<<8192, 192, 0, stream>>>(t2, conv3_w, conv3_b, pooled);
  se_kernel<<<1, 192, 0, stream>>>(pooled, se_w1, se_w2, wgt);
  lce_final_kernel<<<8192, 192, 0, stream>>>(t2, conv3_w, conv3_b, wgt,
                                             (float*)d_out);
}
